// Mamba_61933428417021
// MI455X (gfx1250) — compile-verified, matrix operand order checked by reading, not run on hardware
//
#include <hip/hip_runtime.h>

// ---------------------------------------------------------------------------
// Mamba forward for MI455X (gfx1250): bf16 WMMA GEMMs (double-buffered LDS
// staging) + chunk-parallel selective scan (linear-recurrence decomposition).
// All GEMM shapes padded to M%128 / N%128 / K%64 == 0  ->  guard-free kernel.
// ---------------------------------------------------------------------------

typedef __attribute__((ext_vector_type(16))) __bf16 v16bf;
typedef __attribute__((ext_vector_type(8)))  float  v8f;
typedef __attribute__((ext_vector_type(4)))  float  v4f;

constexpr int BATCH = 2;
constexpr int LSEQ  = 1024;
constexpr int TOK   = BATCH * LSEQ;   // 2048
constexpr int DM    = 768;
constexpr int DI    = 1536;           // d_inner
constexpr int DPROJ = 2 * DI;         // 3072 (in_proj out)
constexpr int VOC   = 32000;
constexpr int DTR   = 48;             // dt_rank
constexpr int NS    = 16;             // d_state
constexpr int XDBL  = DTR + 2 * NS;   // 80 (logical)
constexpr int XDBLP = 128;            // padded x_dbl row stride
constexpr int DTRP  = 64;             // padded dt_rank (K of dt GEMM)
constexpr int CH    = 64;             // scan chunk length
constexpr int NCH   = LSEQ / CH;      // 16 chunks

// ---------------------------------------------------------------------------
// Tiled GEMM: C[M,N] = A[M,K] * Wt[N,K]^T   (A, Wt bf16 row-major, C f32)
// Requires M%128==0, N%128==0, K%64==0 (caller pads). No bounds checks.
// epi: 0 = none, 1 = softplus(C + bias[n]), 2 = C + res[m,n] (residual)
// 128x128 tile, BK=64 (two WMMA K-steps / stage), register double-buffering:
// next global tile is in flight while the current LDS tile feeds 16 WMMAs.
// ---------------------------------------------------------------------------
#define BMT 128
#define BNT 128
#define BKT 64
#define LDT 72   // padded LDS row stride in bf16 elements (144B, 16B-aligned)

__global__ __launch_bounds__(256)
void gemm_bf16_k(const __bf16* __restrict__ A, int lda,
                 const __bf16* __restrict__ Bt, int ldb,
                 float* __restrict__ C, int ldc,
                 int M, int N, int K,
                 const float* __restrict__ bias,
                 const float* __restrict__ res,
                 int epi)
{
    __shared__ __bf16 As[BMT * LDT];
    __shared__ __bf16 Bs[BNT * LDT];

    const int tid    = threadIdx.x;
    const int lane   = tid & 31;
    const int wave   = tid >> 5;
    const int blockM = blockIdx.y * BMT;
    const int blockN = blockIdx.x * BNT;
    const int waveM  = (wave >> 2) * 64;   // 2 wave-rows of 64
    const int waveN  = (wave & 3) * 32;    // 4 wave-cols of 32

    v8f acc[4][2];
    const v8f vzero = {0.f, 0.f, 0.f, 0.f, 0.f, 0.f, 0.f, 0.f};
#pragma unroll
    for (int mt = 0; mt < 4; ++mt)
#pragma unroll
        for (int nt = 0; nt < 2; ++nt) acc[mt][nt] = vzero;

    // tile loader mapping: 2 threads per row, 32 bf16 (64B) each
    const int lrow = tid >> 1;
    const int koff = (tid & 1) * 32;

    // fragment mapping per ISA 16-bit A/B layout (wave32):
    // lanes 0-15: K half 0-7 & 16-23 ; lanes 16-31: K half 8-15 & 24-31
    const int frow  = lane & 15;
    const int khalf = (lane >> 4) * 8;

    const __bf16* aptr = A + (size_t)(blockM + lrow) * lda + koff;
    const __bf16* bptr = Bt + (size_t)(blockN + lrow) * ldb + koff;
    v4f* adst = (v4f*)&As[lrow * LDT + koff];
    v4f* bdst = (v4f*)&Bs[lrow * LDT + koff];

    // prologue: first tile in flight
    v4f abuf[4], bbuf[4];
    {
        const v4f* sa = (const v4f*)aptr;
        const v4f* sb = (const v4f*)bptr;
        abuf[0] = sa[0]; abuf[1] = sa[1]; abuf[2] = sa[2]; abuf[3] = sa[3];
        bbuf[0] = sb[0]; bbuf[1] = sb[1]; bbuf[2] = sb[2]; bbuf[3] = sb[3];
    }

    for (int kk = 0; kk < K; kk += BKT) {
        __syncthreads();   // previous stage done reading LDS
        adst[0] = abuf[0]; adst[1] = abuf[1]; adst[2] = abuf[2]; adst[3] = abuf[3];
        bdst[0] = bbuf[0]; bdst[1] = bbuf[1]; bdst[2] = bbuf[2]; bdst[3] = bbuf[3];
        __syncthreads();

        // kick off next tile's global loads; consumed at next stage's store
        if (kk + BKT < K) {
            const v4f* sa = (const v4f*)(aptr + kk + BKT);
            const v4f* sb = (const v4f*)(bptr + kk + BKT);
            abuf[0] = sa[0]; abuf[1] = sa[1]; abuf[2] = sa[2]; abuf[3] = sa[3];
            bbuf[0] = sb[0]; bbuf[1] = sb[1]; bbuf[2] = sb[2]; bbuf[3] = sb[3];
        }

#pragma unroll
        for (int ks = 0; ks < BKT; ks += 32) {
            v16bf afrag[4], bfrag[2];
#pragma unroll
            for (int mt = 0; mt < 4; ++mt) {
                const __bf16* p = &As[(waveM + mt * 16 + frow) * LDT + ks];
                v4f* d = reinterpret_cast<v4f*>(&afrag[mt]);
                d[0] = *(const v4f*)(p + khalf);        // K = khalf..khalf+7
                d[1] = *(const v4f*)(p + 16 + khalf);   // K = 16+khalf..+7
            }
#pragma unroll
            for (int nt = 0; nt < 2; ++nt) {
                const __bf16* p = &Bs[(waveN + nt * 16 + frow) * LDT + ks];
                v4f* d = reinterpret_cast<v4f*>(&bfrag[nt]);
                d[0] = *(const v4f*)(p + khalf);
                d[1] = *(const v4f*)(p + 16 + khalf);
            }
#pragma unroll
            for (int mt = 0; mt < 4; ++mt)
#pragma unroll
                for (int nt = 0; nt < 2; ++nt)
                    acc[mt][nt] = __builtin_amdgcn_wmma_f32_16x16x32_bf16(
                        false, afrag[mt], false, bfrag[nt],
                        (short)0, acc[mt][nt], false, false);
        }
    }

    // epilogue: C/D layout — lane L, vgpr v: row = v + (L>=16 ? 8 : 0), col = L&15
    const int rowoff = (lane >> 4) * 8;
    const int coloff = lane & 15;
#pragma unroll
    for (int mt = 0; mt < 4; ++mt)
#pragma unroll
        for (int nt = 0; nt < 2; ++nt)
#pragma unroll
            for (int v = 0; v < 8; ++v) {
                const int gm = blockM + waveM + mt * 16 + v + rowoff;
                const int gn = blockN + waveN + nt * 16 + coloff;
                float val = acc[mt][nt][v];
                if (epi == 1) {
                    val += bias[gn];
                    val = (val > 20.f) ? val : log1pf(__expf(val));  // softplus
                } else if (epi == 2) {
                    val += res[(size_t)gm * ldc + gn];
                }
                C[(size_t)gm * ldc + gn] = val;
            }
}

// ---------------------------------------------------------------------------
// Embedding gather: x[t,:] = emb[ids[t],:]
// ---------------------------------------------------------------------------
__global__ __launch_bounds__(256)
void embed_k(const int* __restrict__ ids, const float* __restrict__ emb,
             float* __restrict__ x)
{
    const int t  = blockIdx.x;
    const int id = ids[t];
    for (int i = threadIdx.x; i < DM; i += 256)
        x[(size_t)t * DM + i] = emb[(size_t)id * DM + i];
}

// ---------------------------------------------------------------------------
// RMSNorm over last dim, emits bf16 (feeds the next GEMM's A operand)
// ---------------------------------------------------------------------------
__global__ __launch_bounds__(256)
void rmsnorm_k(const float* __restrict__ x, const float* __restrict__ w,
               __bf16* __restrict__ out, int D)
{
    __shared__ float red[256];
    const int row = blockIdx.x;
    const float* xp = x + (size_t)row * D;
    float ss = 0.f;
    for (int i = threadIdx.x; i < D; i += 256) { float v = xp[i]; ss += v * v; }
    red[threadIdx.x] = ss;
    __syncthreads();
    for (int s = 128; s > 0; s >>= 1) {
        if (threadIdx.x < s) red[threadIdx.x] += red[threadIdx.x + s];
        __syncthreads();
    }
    const float scale = rsqrtf(red[0] / (float)D + 1e-5f);
    for (int i = threadIdx.x; i < D; i += 256)
        out[(size_t)row * D + i] = (__bf16)(xp[i] * scale * w[i]);
}

// ---------------------------------------------------------------------------
// f32 -> bf16 convert (weights, activations)
// ---------------------------------------------------------------------------
__global__ __launch_bounds__(256)
void cvt_k(const float* __restrict__ src, __bf16* __restrict__ dst, size_t n)
{
    size_t i = (size_t)blockIdx.x * blockDim.x + threadIdx.x;
    const size_t stride = (size_t)gridDim.x * blockDim.x;
    for (; i < n; i += stride) dst[i] = (__bf16)src[i];
}

// ---------------------------------------------------------------------------
// f32 -> bf16 convert with zero padding: dst is (R, dld); copies the
// (rows_copy x cols_copy) region of src (row stride sld), zero elsewhere.
// ---------------------------------------------------------------------------
__global__ __launch_bounds__(256)
void cvt_pad_k(const float* __restrict__ src, int sld,
               __bf16* __restrict__ dst, int dld,
               int rows_copy, int cols_copy, size_t total)
{
    size_t i = (size_t)blockIdx.x * blockDim.x + threadIdx.x;
    const size_t stride = (size_t)gridDim.x * blockDim.x;
    for (; i < total; i += stride) {
        const int r = (int)(i / dld);
        const int c = (int)(i % dld);
        const float v = (r < rows_copy && c < cols_copy)
                            ? src[(size_t)r * sld + c] : 0.f;
        dst[i] = (__bf16)v;
    }
}

// ---------------------------------------------------------------------------
// Depthwise causal conv (d_conv=4) + SiLU.  xr is (tok, 3072): u-part cols
// [0,1536), res-part cols [1536,3072). Emits u in f32 (scan) + bf16 (x_proj).
// ---------------------------------------------------------------------------
__global__ __launch_bounds__(256)
void conv_silu_k(const float* __restrict__ xr, const float* __restrict__ cw,
                 const float* __restrict__ cb, float* __restrict__ uf,
                 __bf16* __restrict__ ub)
{
    const int idx = blockIdx.x * 256 + threadIdx.x;   // < TOK*DI
    const int c   = idx % DI;
    const int pos = idx / DI;
    const int l   = pos % LSEQ;
    float acc = cb[c];
#pragma unroll
    for (int j = 0; j < 4; ++j) {
        const int ls = l - 3 + j;
        if (ls >= 0)
            acc = fmaf(cw[c * 4 + j], xr[(size_t)(pos - 3 + j) * DPROJ + c], acc);
    }
    const float s = acc / (1.f + __expf(-acc));       // silu
    uf[idx] = s;
    ub[idx] = (__bf16)s;
}

// ---------------------------------------------------------------------------
// Chunk-parallel selective scan. Recurrence h_t = dA_t*h_{t-1} + dBu_t is
// linear, so L=1024 splits into NCH=16 chunks of CH=64:
//   pass1: per-chunk local scan from h=0; chunk decay P_n = exp(a_n * sum dt)
//   pass2: serial carry over 16 chunks (cheap): H_j = P_{j-1} H_{j-1} + hend
//   pass3: re-scan each chunk seeded with H_j, emit gated bf16 y
// One thread per (b, chunk, c); B_t/C_t shared across the wave via __shfl.
// State buffers layout: [b][chunk][n][c]  (c-contiguous -> coalesced).
// ---------------------------------------------------------------------------
__global__ __launch_bounds__(256)
void scan_chunk_k(const float* __restrict__ delta, const float* __restrict__ u,
                  const float* __restrict__ xdbl, const float* __restrict__ Alog,
                  float* __restrict__ hend, float* __restrict__ pend)
{
    const int gid   = blockIdx.x * 256 + threadIdx.x;  // < BATCH*NCH*DI
    const int c     = gid % DI;
    const int rest  = gid / DI;
    const int chunk = rest % NCH;
    const int b     = rest / NCH;
    const int lane  = threadIdx.x & 31;

    float a[NS], h[NS];
#pragma unroll
    for (int n = 0; n < NS; ++n) {
        a[n] = -__expf(Alog[(size_t)c * NS + n]);
        h[n] = 0.f;
    }
    float sdt = 0.f;
    const int t0 = chunk * CH;
    for (int t = t0; t < t0 + CH; ++t) {
        const size_t r = (size_t)b * LSEQ + t;
        const float bc = xdbl[r * XDBLP + DTR + lane];  // lanes 0-15: B
        const float dt = delta[r * DI + c];
        const float du = dt * u[r * DI + c];
        sdt += dt;
#pragma unroll
        for (int n = 0; n < NS; ++n) {
            const float Bn = __shfl(bc, n, 32);
            h[n] = __expf(dt * a[n]) * h[n] + du * Bn;
        }
    }
#pragma unroll
    for (int n = 0; n < NS; ++n) {
        const size_t idx = ((size_t)(b * NCH + chunk) * NS + n) * DI + c;
        hend[idx] = h[n];
        pend[idx] = __expf(a[n] * sdt);
    }
}

__global__ __launch_bounds__(256)
void scan_carry_k(const float* __restrict__ hend, const float* __restrict__ pend,
                  float* __restrict__ hinit)
{
    const int gid = blockIdx.x * 256 + threadIdx.x;    // < BATCH*NS*DI
    const int c = gid % DI;
    const int n = (gid / DI) % NS;
    const int b = gid / (DI * NS);
    float H = 0.f;
    for (int j = 0; j < NCH; ++j) {
        const size_t idx = ((size_t)(b * NCH + j) * NS + n) * DI + c;
        hinit[idx] = H;
        H = pend[idx] * H + hend[idx];
    }
}

__global__ __launch_bounds__(256)
void scan_out_k(const float* __restrict__ delta, const float* __restrict__ u,
                const float* __restrict__ xdbl, const float* __restrict__ xr,
                const float* __restrict__ Alog, const float* __restrict__ Dv,
                const float* __restrict__ hinit, __bf16* __restrict__ yout)
{
    const int gid   = blockIdx.x * 256 + threadIdx.x;  // < BATCH*NCH*DI
    const int c     = gid % DI;
    const int rest  = gid / DI;
    const int chunk = rest % NCH;
    const int b     = rest / NCH;
    const int lane  = threadIdx.x & 31;

    float a[NS], h[NS];
#pragma unroll
    for (int n = 0; n < NS; ++n) {
        a[n] = -__expf(Alog[(size_t)c * NS + n]);
        h[n] = hinit[((size_t)(b * NCH + chunk) * NS + n) * DI + c];
    }
    const float Dc = Dv[c];

    const int t0 = chunk * CH;
    for (int t = t0; t < t0 + CH; ++t) {
        const size_t r = (size_t)b * LSEQ + t;
        // lanes 0..15 load B[0..15], lanes 16..31 load C[0..15]
        const float bc = xdbl[r * XDBLP + DTR + lane];
        const float dt = delta[r * DI + c];
        const float uv = u[r * DI + c];
        const float du = dt * uv;
        float y = 0.f;
#pragma unroll
        for (int n = 0; n < NS; ++n) {
            const float Bn = __shfl(bc, n, 32);
            const float Cn = __shfl(bc, 16 + n, 32);
            h[n] = __expf(dt * a[n]) * h[n] + du * Bn;
            y = fmaf(h[n], Cn, y);
        }
        y = fmaf(uv, Dc, y);
        const float rg = xr[r * DPROJ + DI + c];
        y *= rg / (1.f + __expf(-rg));          // * silu(res)
        yout[r * DI + c] = (__bf16)y;
    }
}

// ---------------------------------------------------------------------------
// Host orchestration (graph-capture safe: only kernel launches on `stream`)
// ---------------------------------------------------------------------------
extern "C" void kernel_launch(void* const* d_in, const int* in_sizes, int n_in,
                              void* d_out, int out_size, void* d_ws, size_t ws_size,
                              hipStream_t stream)
{
    (void)in_sizes; (void)n_in; (void)out_size; (void)ws_size;

    const int*   ids     = (const int*)  d_in[0];
    const float* emb     = (const float*)d_in[1];
    const float* norm_w  = (const float*)d_in[2];
    const float* in_w    = (const float*)d_in[3];
    const float* conv_w  = (const float*)d_in[4];
    const float* conv_b  = (const float*)d_in[5];
    const float* xp_w    = (const float*)d_in[6];
    const float* dt_w    = (const float*)d_in[7];
    const float* dt_b    = (const float*)d_in[8];
    const float* A_log   = (const float*)d_in[9];
    const float* Dvec    = (const float*)d_in[10];
    const float* out_w   = (const float*)d_in[11];
    const float* normf_w = (const float*)d_in[12];
    float* out = (float*)d_out;

    char* ws = (char*)d_ws;
    size_t off = 0;
    auto carve = [&](size_t bytes) -> void* {
        void* p = ws + off;
        off += (bytes + 255) & ~(size_t)255;
        return p;
    };
    constexpr size_t NSTATE = (size_t)BATCH * NCH * NS * DI;  // 786432

    float*  x      = (float*) carve((size_t)TOK * DM * 4);
    __bf16* xn     = (__bf16*)carve((size_t)TOK * DM * 2);
    float*  xr     = (float*) carve((size_t)TOK * DPROJ * 4);
    float*  uf     = (float*) carve((size_t)TOK * DI * 4);
    __bf16* ub     = (__bf16*)carve((size_t)TOK * DI * 2);
    float*  xdbl   = (float*) carve((size_t)TOK * XDBLP * 4);   // padded 2048x128
    __bf16* xdt    = (__bf16*)carve((size_t)TOK * DTRP * 2);    // padded 2048x64
    float*  delta  = (float*) carve((size_t)TOK * DI * 4);
    __bf16* yb     = (__bf16*)carve((size_t)TOK * DI * 2);
    float*  hend   = (float*) carve(NSTATE * 4);
    float*  pend   = (float*) carve(NSTATE * 4);
    float*  hinit  = (float*) carve(NSTATE * 4);
    __bf16* w_in   = (__bf16*)carve((size_t)DPROJ * DM * 2);
    __bf16* w_xp   = (__bf16*)carve((size_t)XDBLP * DI * 2);    // padded 128x1536
    __bf16* w_dt   = (__bf16*)carve((size_t)DI * DTRP * 2);     // padded 1536x64
    __bf16* w_out  = (__bf16*)carve((size_t)DM * DI * 2);
    __bf16* embb   = (__bf16*)carve((size_t)VOC * DM * 2);

    embed_k<<<TOK, 256, 0, stream>>>(ids, emb, x);

    for (int i = 0; i < 2; ++i) {
        // rmsnorm -> bf16 operand
        rmsnorm_k<<<TOK, 256, 0, stream>>>(x, norm_w + (size_t)i * DM, xn, DM);

        // in_proj: (2048,768) x (3072,768)^T -> xr (2048,3072)
        cvt_k<<<4096, 256, 0, stream>>>(in_w + (size_t)i * DPROJ * DM, w_in,
                                        (size_t)DPROJ * DM);
        gemm_bf16_k<<<dim3(DPROJ / BNT, TOK / BMT), 256, 0, stream>>>(
            xn, DM, w_in, DM, xr, DPROJ, TOK, DPROJ, DM, nullptr, nullptr, 0);

        // causal depthwise conv + silu -> u
        conv_silu_k<<<(TOK * DI) / 256, 256, 0, stream>>>(
            xr, conv_w + (size_t)i * DI * 4, conv_b + (size_t)i * DI, uf, ub);

        // x_proj: (2048,1536) x (128pad,1536)^T -> x_dbl (2048,128pad)
        cvt_pad_k<<<768, 256, 0, stream>>>(xp_w + (size_t)i * XDBL * DI, DI,
                                           w_xp, DI, XDBL, DI,
                                           (size_t)XDBLP * DI);
        gemm_bf16_k<<<dim3(XDBLP / BNT, TOK / BMT), 256, 0, stream>>>(
            ub, DI, w_xp, DI, xdbl, XDBLP, TOK, XDBLP, DI, nullptr, nullptr, 0);

        // delta = softplus(x_dbl[:, :48] x dt_proj^T + b): K padded 48 -> 64
        cvt_pad_k<<<512, 256, 0, stream>>>(xdbl, XDBLP, xdt, DTRP,
                                           TOK, DTR, (size_t)TOK * DTRP);
        cvt_pad_k<<<384, 256, 0, stream>>>(dt_w + (size_t)i * DI * DTR, DTR,
                                           w_dt, DTRP, DI, DTR,
                                           (size_t)DI * DTRP);
        gemm_bf16_k<<<dim3(DI / BNT, TOK / BMT), 256, 0, stream>>>(
            xdt, DTRP, w_dt, DTRP, delta, DI, TOK, DI, DTRP,
            dt_b + (size_t)i * DI, nullptr, 1);

        // chunk-parallel selective scan (+ gating) -> y bf16
        scan_chunk_k<<<(BATCH * NCH * DI) / 256, 256, 0, stream>>>(
            delta, uf, xdbl, A_log + (size_t)i * DI * NS, hend, pend);
        scan_carry_k<<<(BATCH * NS * DI) / 256, 256, 0, stream>>>(
            hend, pend, hinit);
        scan_out_k<<<(BATCH * NCH * DI) / 256, 256, 0, stream>>>(
            delta, uf, xdbl, xr, A_log + (size_t)i * DI * NS,
            Dvec + (size_t)i * DI, hinit, yb);

        // out_proj + residual: x = x + y x (768,1536)^T
        cvt_k<<<4096, 256, 0, stream>>>(out_w + (size_t)i * DM * DI, w_out,
                                        (size_t)DM * DI);
        gemm_bf16_k<<<dim3(DM / BNT, TOK / BMT), 256, 0, stream>>>(
            yb, DI, w_out, DI, x, DM, TOK, DM, DI, nullptr, x, 2);
    }

    // final norm + tied lm_head: (2048,768) x (32000,768)^T -> logits
    rmsnorm_k<<<TOK, 256, 0, stream>>>(x, normf_w, xn, DM);
    cvt_k<<<8192, 256, 0, stream>>>(emb, embb, (size_t)VOC * DM);
    gemm_bf16_k<<<dim3(VOC / BNT, TOK / BMT), 256, 0, stream>>>(
        xn, DM, embb, DM, out, VOC, TOK, VOC, DM, nullptr, nullptr, 0);
}
